// CENTER_HEAD_35390530519358
// MI455X (gfx1250) — matrix operand order repaired and checked by reading, not compiled, not run on hardware
//
#include <hip/hip_runtime.h>
#include <hip/hip_bf16.h>
#include <math.h>

// ---------------- problem constants ----------------
#define Bn   4
#define CIN  256
#define EMBD 256
#define Hd   120
#define Wd   200
#define HWd  (Hd * Wd)       // 24000
#define Pn   200
#define DSF  16

// GEMM tiling: block = 256 threads = 8 waves; tile 128(M) x 64(N), K-step 32.
#define TM 128
#define TN 64
#define FS 40                // per-row LDS stride in bf16 (80B -> bank step 20)

static_assert(HWd  % TN == 0, "pixel count must tile by TN");
static_assert(EMBD % TM == 0, "channels must tile by TM");

typedef __attribute__((ext_vector_type(16))) __bf16 v16bf;
typedef __attribute__((ext_vector_type(8)))  __bf16 v8bf;
typedef __attribute__((ext_vector_type(8)))  float  v8f;

// ---- LDS fragment-order layout -------------------------------------------
// A tile (TM x 32, MxK): row m holds 32 bf16 at positions
//   pos(k) = ((k>>3)&1)*16 + (k>>4)*8 + (k&7)
// so a lane (m = mo+(L&15), hs = L>>4) reads its whole 16-elem WMMA A
// fragment (per CDNA5 ISA: elems 0-7 -> k=hs*8+0..7,
// elems 8-15 -> k=16+hs*8+0..7) as 32 CONTIGUOUS bytes: As[m*FS + hs*16 ..].
// B tile (32 x TN, KxN): column n holds 32 bf16 at the same pos(k) as A,
// lane (n = nb+(L&15), kg = L>>4) reads Bs[n*FS + kg*16 ..] contiguously.
__device__ __forceinline__ v16bf ld_frag(const __bf16* p) {
  const v8bf lo = *(const v8bf*)(p);
  const v8bf hi = *(const v8bf*)(p + 8);
  return __builtin_shufflevector(lo, hi, 0,1,2,3,4,5,6,7,8,9,10,11,12,13,14,15);
}

__device__ __forceinline__ v8bf pack8(const float* v0, const float* v1) {
  v8bf p;
#pragma unroll
  for (int j = 0; j < 4; ++j) { p[j] = (__bf16)v0[j]; p[j+4] = (__bf16)v1[j]; }
  return p;
}

// ---------------- Kernel 1: 1x1 conv as GEMM (bf16 WMMA, f32 accum) -------
// out[b,o,n] = sum_c w[o,c]*feat[b,c,n] + bias[o]
__global__ __launch_bounds__(256) void k_conv1x1(
    const float* __restrict__ feat, const float* __restrict__ w,
    const float* __restrict__ bias, float* __restrict__ out)
{
  __shared__ alignas(16) __bf16 As[TM * FS];   // 10240 B
  __shared__ alignas(16) __bf16 Bs[TN * FS];   //  5120 B
  const int n0 = blockIdx.x * TN;
  const int o0 = blockIdx.y * TM;
  const int b  = blockIdx.z;
  const int tid = threadIdx.x, lane = tid & 31, wv = tid >> 5;
  const int mo = wv << 4;                      // wave's 16-row M sub-tile
  const int lm = lane & 15, sel = lane >> 4;   // sel = hs (A) = kg (B)

  // B staging task (fixed per thread): 64-thread groups share one k-chunk
  const int bn  = tid & 63;                    // pixel column within tile
  const int bkb = (tid >> 6) * 8;              // k chunk base (0,8,16,24)

  v8f acc[4] = {};

  for (int c0 = 0; c0 < CIN; c0 += 32) {
    // stage A: TMx32 weights, 2 chunk-tasks of 8 consecutive k per thread
#pragma unroll
    for (int e = 0; e < 2; ++e) {
      const int tt = tid + e * 256;
      const int m = tt >> 2, kb = (tt & 3) * 8;
      const float* wr = &w[(size_t)(o0 + m) * CIN + c0 + kb];
      const int pos = ((kb >> 3) & 1) * 16 + (kb >> 4) * 8;
      *(v8bf*)&As[m * FS + pos] = pack8(wr, wr + 4);
    }
    // stage B: 32xTN activations, one 8-k column chunk per thread
    {
      const float* src = &feat[((size_t)b * CIN + c0 + bkb) * HWd + n0 + bn];
      float t0[4], t1[4];
#pragma unroll
      for (int j = 0; j < 4; ++j) { t0[j] = src[(size_t)j * HWd];
                                    t1[j] = src[(size_t)(j + 4) * HWd]; }
      const int pos = ((bkb >> 3) & 1) * 16 + (bkb >> 4) * 8;
      *(v8bf*)&Bs[bn * FS + pos] = pack8(t0, t1);
    }
    __syncthreads();
    const v16bf af = ld_frag(&As[(mo + lm) * FS + sel * 16]);
#pragma unroll
    for (int q = 0; q < 4; ++q) {
      const v16bf bf = ld_frag(&Bs[(q * 16 + lm) * FS + sel * 16]);
      acc[q] = __builtin_amdgcn_wmma_f32_16x16x32_bf16(false, af, false, bf,
                                                       (short)0, acc[q], false, false);
    }
    __syncthreads();
  }

  // C layout: lanes 0-15: N=lane, VGPR r -> M=r ; lanes 16-31: N=lane-16, M=8+r
  const int nq = lane & 15, mb = (lane >> 4) << 3;
#pragma unroll
  for (int r = 0; r < 8; ++r) {
    const int m = o0 + mo + mb + r;
    const float bi = bias[m];
    float* o = out + ((size_t)b * EMBD + m) * HWd + n0 + nq;
#pragma unroll
    for (int q = 0; q < 4; ++q) o[q * 16] = acc[q][r] + bi;
  }
}

// ---------------- Kernel 2/3: 3x3 conv + BN (+residual) + ReLU ------------
// GEMM with K = 9 taps x 256 channels (im2col on the fly, zero padded).
__global__ __launch_bounds__(256) void k_conv3x3_bn(
    const float* __restrict__ in, const float* __restrict__ w9,   // (O,C,3,3)
    const float* __restrict__ scale, const float* __restrict__ shift,
    const float* __restrict__ residual,                            // may be null
    float* __restrict__ out)
{
  __shared__ alignas(16) __bf16 As[TM * FS];
  __shared__ alignas(16) __bf16 Bs[TN * FS];
  const int n0 = blockIdx.x * TN;
  const int o0 = blockIdx.y * TM;
  const int b  = blockIdx.z;
  const int tid = threadIdx.x, lane = tid & 31, wv = tid >> 5;
  const int mo = wv << 4;
  const int lm = lane & 15, sel = lane >> 4;

  // fixed per-thread B staging pixel: hoist the div/mod out of all loops
  const int bn  = tid & 63;
  const int bkb = (tid >> 6) * 8;
  const int bpos = ((bkb >> 3) & 1) * 16 + (bkb >> 4) * 8;
  const int p  = n0 + bn;
  const int hh = p / Wd, ww = p - hh * Wd;

  v8f acc[4] = {};

  for (int t = 0; t < 9; ++t) {
    const int ky = t / 3 - 1, kx = t % 3 - 1;
    const int sh = hh + ky, sw = ww + kx;
    const bool valid = (sh >= 0) & (sh < Hd) & (sw >= 0) & (sw < Wd);
    const float* srcpix = in + (size_t)b * CIN * HWd + sh * Wd + sw;

    for (int c0 = 0; c0 < CIN; c0 += 32) {
      // stage A: weight tap slice (stride 9 over c)
#pragma unroll
      for (int e = 0; e < 2; ++e) {
        const int tt = tid + e * 256;
        const int m = tt >> 2, kb = (tt & 3) * 8;
        const size_t wb = ((size_t)(o0 + m) * CIN + c0 + kb) * 9 + t;
        v8bf pk;
#pragma unroll
        for (int j = 0; j < 8; ++j) pk[j] = (__bf16)w9[wb + (size_t)j * 9];
        const int pos = ((kb >> 3) & 1) * 16 + (kb >> 4) * 8;
        *(v8bf*)&As[m * FS + pos] = pk;
      }
      // stage B: shifted activations, zero padded
      {
        const float* src = srcpix + (size_t)(c0 + bkb) * HWd;
        v8bf pk;
#pragma unroll
        for (int j = 0; j < 8; ++j)
          pk[j] = (__bf16)(valid ? src[(size_t)j * HWd] : 0.f);
        *(v8bf*)&Bs[bn * FS + bpos] = pk;
      }
      if (valid && c0 + 32 < CIN)   // warm caches for next K-tile
        __builtin_prefetch(srcpix + (size_t)(c0 + 32 + (lane & 31)) * HWd, 0, 1);
      __syncthreads();
      const v16bf af = ld_frag(&As[(mo + lm) * FS + sel * 16]);
#pragma unroll
      for (int q = 0; q < 4; ++q) {
        const v16bf bf = ld_frag(&Bs[(q * 16 + lm) * FS + sel * 16]);
        acc[q] = __builtin_amdgcn_wmma_f32_16x16x32_bf16(false, af, false, bf,
                                                         (short)0, acc[q], false, false);
      }
      __syncthreads();
    }
  }

  const int nq = lane & 15, mb = (lane >> 4) << 3;
#pragma unroll
  for (int r = 0; r < 8; ++r) {
    const int m = o0 + mo + mb + r;
    const float sc = scale[m], sb = shift[m];
    const size_t base = ((size_t)b * EMBD + m) * HWd + n0 + nq;
#pragma unroll
    for (int q = 0; q < 4; ++q) {
      float v = acc[q][r] * sc + sb;
      if (residual) v += residual[base + q * 16];
      out[base + q * 16] = fmaxf(v, 0.f);
    }
  }
}

// ---------------- Kernel 4: prediction heads ------------------------------
__global__ __launch_bounds__(256) void k_heads(
    const float* __restrict__ x,
    const float* __restrict__ w_obj, const float* __restrict__ b_obj,
    const float* __restrict__ w_dep, const float* __restrict__ b_dep,
    const float* __restrict__ w_off, const float* __restrict__ b_off,
    float* __restrict__ obj_out, float* __restrict__ depth_map,
    float* __restrict__ off_map)
{
  const int n = blockIdx.x * 256 + threadIdx.x;
  const int b = blockIdx.y;
  if (n >= HWd) return;
  float a0 = 0.f, a1 = 0.f, a2 = 0.f, a3 = 0.f;
  const float* xb = x + (size_t)b * EMBD * HWd + n;
#pragma unroll 4
  for (int c = 0; c < EMBD; ++c) {
    const float v = xb[(size_t)c * HWd];
    a0 = fmaf(v, w_obj[c], a0);
    a1 = fmaf(v, w_dep[c], a1);
    a2 = fmaf(v, w_off[c], a2);
    a3 = fmaf(v, w_off[EMBD + c], a3);
  }
  const float zo = a0 + b_obj[0];
  float obj = 1.f / (1.f + __expf(-zo));
  obj = fminf(fmaxf(obj, 1e-4f), 1.f - 1e-4f);
  const float zd = a1 + b_dep[0];
  obj_out[(size_t)b * HWd + n]   = obj;
  depth_map[(size_t)b * HWd + n] = __expf(-zd);   // 1/sigmoid(z) - 1 == exp(-z)
  off_map[((size_t)b * 2 + 0) * HWd + n] = a2 + b_off[0];
  off_map[((size_t)b * 2 + 1) * HWd + n] = a3 + b_off[1];
}

// ---------------- Kernel 5: top-k (one block per batch) -------------------
__global__ __launch_bounds__(256) void k_topk(
    const float* __restrict__ obj, float* __restrict__ scratch,
    float* __restrict__ conf_out, int* __restrict__ idx_out)
{
  const int b = blockIdx.x, tid = threadIdx.x;
  float* s = scratch + (size_t)b * HWd;
  for (int i = tid; i < HWd; i += 256) s[i] = obj[(size_t)b * HWd + i];
  __syncthreads();
  __shared__ float sv[256];
  __shared__ int   si[256];
  for (int sel = 0; sel < Pn; ++sel) {
    float best = -1e30f; int bi = 0x7fffffff;
    for (int i = tid; i < HWd; i += 256) {
      const float v = s[i];
      if (v > best || (v == best && i < bi)) { best = v; bi = i; }
    }
    sv[tid] = best; si[tid] = bi;
    __syncthreads();
    for (int off = 128; off > 0; off >>= 1) {
      if (tid < off) {
        const float v2 = sv[tid + off]; const int i2 = si[tid + off];
        if (v2 > sv[tid] || (v2 == sv[tid] && i2 < si[tid])) { sv[tid] = v2; si[tid] = i2; }
      }
      __syncthreads();
    }
    if (tid == 0) {
      conf_out[b * Pn + sel] = sv[0];
      idx_out[b * Pn + sel]  = si[0];
      s[si[0]] = -1e30f;     // knock out winner for next round
    }
    __syncthreads();
  }
}

// ---------------- Kernel 6: bilinear sample + projection ------------------
__device__ __forceinline__ float bilin(const float* __restrict__ plane,
                                       float gx, float gy)
{
  const float x = ((gx + 1.f) * (float)Wd - 1.f) * 0.5f;
  const float y = ((gy + 1.f) * (float)Hd - 1.f) * 0.5f;
  const float x0 = floorf(x), y0 = floorf(y);
  const float wx1 = x - x0, wx0 = 1.f - wx1;
  const float wy1 = y - y0, wy0 = 1.f - wy1;
  const int ix0 = (int)x0, iy0 = (int)y0;
  float acc = 0.f;
#pragma unroll
  for (int dy = 0; dy < 2; ++dy)
#pragma unroll
    for (int dx = 0; dx < 2; ++dx) {
      const int ix = ix0 + dx, iy = iy0 + dy;
      if (ix >= 0 && ix < Wd && iy >= 0 && iy < Hd)
        acc += plane[iy * Wd + ix] * (dx ? wx1 : wx0) * (dy ? wy1 : wy0);
    }
  return acc;
}

__global__ __launch_bounds__(256) void k_project(
    const float* __restrict__ Ks,
    const float* __restrict__ depth_map, const float* __restrict__ off_map,
    const float* __restrict__ conf_in, const int* __restrict__ idx_in,
    float* __restrict__ conf_out, float* __restrict__ xyz_out)
{
  const int g = blockIdx.x * blockDim.x + threadIdx.x;
  if (g >= Bn * Pn) return;
  const int b = g / Pn;
  const int idx = idx_in[g];
  const float u = (float)(idx % Wd);
  const float v = (float)(idx / Hd);                // faithful: // feat_h
  const float gu = (u - Wd * 0.5f) / (Wd * 0.5f);
  const float gv = (v - v * 0.5f) / (Hd * 0.5f);    // faithful: (v - v/2)/(H/2)
  const float* dp  = depth_map + (size_t)b * HWd;
  const float* o0p = off_map + ((size_t)b * 2 + 0) * HWd;
  const float* o1p = off_map + ((size_t)b * 2 + 1) * HWd;
  const float d  = bilin(dp,  gu, gv);
  const float ox = bilin(o0p, gu, gv);
  const float oy = bilin(o1p, gu, gv);
  const float X = (u + ox) * (float)DSF * d;
  const float Y = (v + oy) * (float)DSF * d;
  const float Z = d;
  const float* K = Ks + b * 9;
  const float a00=K[0],a01=K[1],a02=K[2],a10=K[3],a11=K[4],a12=K[5],a20=K[6],a21=K[7],a22=K[8];
  const float det = a00*(a11*a22 - a12*a21) - a01*(a10*a22 - a12*a20) + a02*(a10*a21 - a11*a20);
  const float inv = 1.f / det;
  const float i00 =  (a11*a22 - a12*a21)*inv, i01 = -(a01*a22 - a02*a21)*inv, i02 =  (a01*a12 - a02*a11)*inv;
  const float i10 = -(a10*a22 - a12*a20)*inv, i11 =  (a00*a22 - a02*a20)*inv, i12 = -(a00*a12 - a02*a10)*inv;
  const float i20 =  (a10*a21 - a11*a20)*inv, i21 = -(a00*a21 - a01*a20)*inv, i22 =  (a00*a11 - a01*a10)*inv;
  conf_out[g] = conf_in[g];
  xyz_out[g * 3 + 0] = i00*X + i01*Y + i02*Z;
  xyz_out[g * 3 + 1] = i10*X + i11*Y + i12*Z;
  xyz_out[g * 3 + 2] = i20*X + i21*Y + i22*Z;
}

// ---------------- launcher ------------------------------------------------
extern "C" void kernel_launch(void* const* d_in, const int* in_sizes, int n_in,
                              void* d_out, int out_size, void* d_ws, size_t ws_size,
                              hipStream_t stream)
{
  const float* feat  = (const float*)d_in[0];
  const float* Ks    = (const float*)d_in[1];
  const float* w_sc  = (const float*)d_in[2];
  const float* b_sc  = (const float*)d_in[3];
  const float* w_bb1 = (const float*)d_in[4];
  const float* bn1s  = (const float*)d_in[5];
  const float* bn1b  = (const float*)d_in[6];
  const float* w_bb2 = (const float*)d_in[7];
  const float* bn2s  = (const float*)d_in[8];
  const float* bn2b  = (const float*)d_in[9];
  const float* w_obj = (const float*)d_in[10];
  const float* b_obj = (const float*)d_in[11];
  const float* w_dep = (const float*)d_in[12];
  const float* b_dep = (const float*)d_in[13];
  const float* w_off = (const float*)d_in[14];
  const float* b_off = (const float*)d_in[15];

  float* ws = (float*)d_ws;
  const size_t plane = (size_t)Bn * EMBD * HWd;   // 24,576,000 floats
  float* x0        = ws;                          // conv1x1 output (residual)
  float* h1        = x0 + plane;                  // first basic-block stage
  float* x2        = h1 + plane;                  // block output
  float* depth_map = x2 + plane;                  // (B,1,H,W)
  float* off_map   = depth_map + (size_t)Bn * HWd;       // (B,2,H,W)
  float* obj_scr   = off_map + (size_t)Bn * 2 * HWd;     // top-k scratch copy
  float* conf_ws   = obj_scr + (size_t)Bn * HWd;         // (B,P)
  int*   idx_ws    = (int*)(conf_ws + Bn * Pn);          // (B,P)

  float* obj_out  = (float*)d_out;                // (B,1,H,W)  -> 96000
  float* conf_out = obj_out + (size_t)Bn * HWd;   // (B,P,1)    -> 800
  float* xyz_out  = conf_out + Bn * Pn;           // (B,P,3)    -> 2400

  const dim3 gemm_grid(HWd / TN, EMBD / TM, Bn);
  k_conv1x1<<<gemm_grid, 256, 0, stream>>>(feat, w_sc, b_sc, x0);
  k_conv3x3_bn<<<gemm_grid, 256, 0, stream>>>(x0, w_bb1, bn1s, bn1b, nullptr, h1);
  k_conv3x3_bn<<<gemm_grid, 256, 0, stream>>>(h1, w_bb2, bn2s, bn2b, x0, x2);
  k_heads<<<dim3((HWd + 255) / 256, Bn), 256, 0, stream>>>(
      x2, w_obj, b_obj, w_dep, b_dep, w_off, b_off, obj_out, depth_map, off_map);
  k_topk<<<Bn, 256, 0, stream>>>(obj_out, obj_scr, conf_ws, idx_ws);
  k_project<<<(Bn * Pn + 255) / 256, 256, 0, stream>>>(
      Ks, depth_map, off_map, conf_ws, idx_ws, conf_out, xyz_out);
}
